// MinimalBipartiteGNN_31413390803034
// MI455X (gfx1250) — matrix operand order repaired and checked by reading, not compiled, not run on hardware
//
#include <hip/hip_runtime.h>
#include <hip/hip_bf16.h>

// ---------------------------------------------------------------------------
// CDNA5 (gfx1250) WMMA helpers: D = A(16x32 f16) x B(32x16 f16) + C(16x16 f32)
// ---------------------------------------------------------------------------
typedef __attribute__((ext_vector_type(16))) _Float16 v16h;
typedef __attribute__((ext_vector_type(8)))  _Float16 v8h;
typedef __attribute__((ext_vector_type(8)))  float    v8f;
typedef __attribute__((ext_vector_type(4)))  int      v4i;

#if defined(__gfx1250__) && __has_builtin(__builtin_amdgcn_global_load_async_to_lds_b128)
#define ASYNC_LDS 1
#else
#define ASYNC_LDS 0
#endif

// A operand, 16-bit 16x32 layout (ISA 7.12.2):
//  lanes 0-15 : M=lane,    VGPR0-3 = K0..7,  VGPR4-7 = K16..23
//  lanes16-31 : M=lane-16, VGPR0-3 = K8..15, VGPR4-7 = K24..31
__device__ __forceinline__ v16h wmma_load_a(const _Float16* base, int kstride,
                                            int lane, int k0) {
  const _Float16* row = base + (size_t)(lane & 15) * kstride + k0;
  const int lo = (lane < 16) ? 0 : 8;
  union { v16h v; v8h h[2]; } u;
  u.h[0] = *(const v8h*)(row + lo);
  u.h[1] = *(const v8h*)(row + lo + 16);
  return u.v;
}

// B operand, 16-bit 32x16 layout: lane n holds col n, same K order as A
// (lanes 0-15: K0..7,K16..23, lanes 16-31: K8..15,K24..31). 'wt' is W^T as [n][kstride] f16.
__device__ __forceinline__ v16h wmma_load_b(const _Float16* wt, int kstride,
                                            int jn, int lane, int k0) {
  const _Float16* row = wt + (size_t)(jn * 16 + (lane & 15)) * kstride + k0 +
                        ((lane < 16) ? 0 : 8);
  union { v16h v; v8h h[2]; } u;
  u.h[0] = *(const v8h*)(row);
  u.h[1] = *(const v8h*)(row + 16);
  return u.v;
}

__device__ __forceinline__ v8f wmma_f16(v16h a, v16h b, v8f c) {
  return __builtin_amdgcn_wmma_f32_16x16x32_f16(false, a, false, b,
                                                (short)0, c, false, false);
}

__device__ __forceinline__ v8f splat8(float v) {
  v8f r = {v, v, v, v, v, v, v, v};
  return r;
}

__device__ __forceinline__ void wait_lds() {
  asm volatile("s_wait_dscnt 0" ::: "memory");
}

// Async global->LDS copy of 16B per lane (GLOBAL_LOAD_ASYNC_TO_LDS_B128,
// ASYNCcnt-tracked). OFF is the immediate byte offset applied to both sides.
// Builtin signature (probe-verified): (v4i AS1*, v4i AS3*, Imm, Imm).
template <int OFF>
__device__ __forceinline__ void async_b128(const void* g, void* l) {
#if ASYNC_LDS
  typedef __attribute__((address_space(1))) v4i* gp_t;
  typedef __attribute__((address_space(3))) v4i* lp_t;
  __builtin_amdgcn_global_load_async_to_lds_b128((gp_t)g, (lp_t)l, OFF, 0);
#else
  (void)g; (void)l;
#endif
}

#if ASYNC_LDS
#define WAIT_ASYNC(n) asm volatile("s_wait_asynccnt " #n ::: "memory")
#else
#define WAIT_ASYNC(n)
#endif

// ---------------------------------------------------------------------------
// Generic 2-layer MLP (relu(relu(X W1 + b1) W2 + b2)), DH=64, DOUT=32,
// optional fused 3rd GEMM (Y2 @ W3 + b3, no relu) producing 64 cols.
// One 16-row tile per wave, 4 waves per block. (Tiny kernels; <0.2% of work.)
// ---------------------------------------------------------------------------
template <int DIN, bool HAS3>
__global__ void __launch_bounds__(128)
k_mlp(const float* __restrict__ X, int x_stride, int n_rows,
      const float* __restrict__ W1, const float* __restrict__ B1,
      const float* __restrict__ W2, const float* __restrict__ B2,
      const float* __restrict__ W3, const float* __restrict__ B3,
      float* __restrict__ Y, int y_stride) {
  constexpr int DH = 64, DOUT = 32, D3 = 64;
  __shared__ __align__(32) _Float16 Wt1[DH * DIN];
  __shared__ __align__(32) _Float16 Wt2[DOUT * DH];
  __shared__ __align__(32) _Float16 Wt3[HAS3 ? D3 * DOUT : 16];
  __shared__ __align__(32) _Float16 Xs[4][16 * DIN];
  __shared__ __align__(32) _Float16 Hs[4][16 * DH];
  __shared__ __align__(32) _Float16 Y2[HAS3 ? 4 : 1][HAS3 ? 16 * DOUT : 16];
  __shared__ float b1s[DH], b2s[DOUT], b3s[D3];

  const int tid = threadIdx.x;
  for (int i = tid; i < DH * DIN; i += 128) {
    int n = i / DIN, k = i % DIN;
    Wt1[i] = (_Float16)W1[k * DH + n];
  }
  for (int i = tid; i < DOUT * DH; i += 128) {
    int n = i / DH, k = i % DH;
    Wt2[i] = (_Float16)W2[k * DOUT + n];
  }
  if constexpr (HAS3) {
    for (int i = tid; i < D3 * DOUT; i += 128) {
      int n = i / DOUT, k = i % DOUT;
      Wt3[i] = (_Float16)W3[k * D3 + n];
    }
    for (int i = tid; i < D3; i += 128) b3s[i] = B3 ? B3[i] : 0.f;
  }
  for (int i = tid; i < DH; i += 128) b1s[i] = B1[i];
  for (int i = tid; i < DOUT; i += 128) b2s[i] = B2[i];
  __syncthreads();

  const int wid = tid >> 5, lane = tid & 31;
  const int col = lane & 15, mrow = (lane >> 4) * 8;
  const int row0 = (blockIdx.x * 4 + wid) * 16;
  if (row0 >= n_rows) return;

  _Float16* xs = Xs[wid];
  _Float16* hs = Hs[wid];
  for (int i = lane; i < 16 * DIN; i += 32) {
    int r = i / DIN, c = i % DIN;
    xs[i] = (_Float16)X[(size_t)(row0 + r) * x_stride + c];
  }
  wait_lds();

  // stage 1: DIN -> 64
#pragma unroll
  for (int jn = 0; jn < 4; ++jn) {
    v8f c = splat8(b1s[jn * 16 + col]);
#pragma unroll
    for (int k0 = 0; k0 < DIN; k0 += 32)
      c = wmma_f16(wmma_load_a(xs, DIN, lane, k0),
                   wmma_load_b(Wt1, DIN, jn, lane, k0), c);
#pragma unroll
    for (int r = 0; r < 8; ++r)
      hs[(mrow + r) * DH + jn * 16 + col] = (_Float16)fmaxf((float)c[r], 0.f);
  }
  wait_lds();

  // stage 2: 64 -> 32
#pragma unroll
  for (int jn = 0; jn < 2; ++jn) {
    v8f c = splat8(b2s[jn * 16 + col]);
    c = wmma_f16(wmma_load_a(hs, DH, lane, 0),
                 wmma_load_b(Wt2, DH, jn, lane, 0), c);
    c = wmma_f16(wmma_load_a(hs, DH, lane, 32),
                 wmma_load_b(Wt2, DH, jn, lane, 32), c);
    if constexpr (!HAS3) {
#pragma unroll
      for (int r = 0; r < 8; ++r)
        Y[(size_t)(row0 + mrow + r) * y_stride + jn * 16 + col] =
            fmaxf((float)c[r], 0.f);
    } else {
#pragma unroll
      for (int r = 0; r < 8; ++r)
        Y2[wid][(mrow + r) * DOUT + jn * 16 + col] =
            (_Float16)fmaxf((float)c[r], 0.f);
    }
  }
  if constexpr (HAS3) {
    wait_lds();
    // stage 3: 32 -> 64, no relu (projection through sh_w1 slice + opt bias)
#pragma unroll
    for (int jn = 0; jn < 4; ++jn) {
      v8f c = splat8(b3s[jn * 16 + col]);
      c = wmma_f16(wmma_load_a(Y2[wid], DOUT, lane, 0),
                   wmma_load_b(Wt3, DOUT, jn, lane, 0), c);
#pragma unroll
      for (int r = 0; r < 8; ++r)
        Y[(size_t)(row0 + mrow + r) * y_stride + jn * 16 + col] = (float)c[r];
    }
  }
}

// ---------------------------------------------------------------------------
// Pair MLP pass: one block per task row; each wave covers 16 server tiles of
// 16 pairs. B operands AND bias C-tiles hoisted to registers (WMMA SRC2 and
// VDST are separate fields, so a live bias tile needs no per-tile copies).
// pair_feat tiles staged with async global->LDS (double buffered).
// p_emb f16 [T][S][32] -> ws (128 MB, L2-resident on MI455X).
// ---------------------------------------------------------------------------
__global__ void __launch_bounds__(128)
k_pair1(const float* __restrict__ PF, const float* __restrict__ W1,
        const float* __restrict__ B1, const float* __restrict__ W2,
        const float* __restrict__ B2, _Float16* __restrict__ PE) {
  __shared__ __align__(32) _Float16 Wt1[64 * 32];  // [n][k], k 16..31 zero pad
  __shared__ __align__(32) _Float16 Wt2[32 * 64];
  __shared__ __align__(32) _Float16 Xs[4][16 * 32];
  __shared__ __align__(32) _Float16 Hs[4][16 * 64];
  __shared__ __align__(16) float    Xf[4][2][256];   // async f32 staging
  __shared__ float b1s[64], b2s[32];

  const int tid = threadIdx.x;
  for (int i = tid; i < 64 * 32; i += 128) {
    int n = i / 32, k = i % 32;
    Wt1[i] = (k < 16) ? (_Float16)W1[k * 64 + n] : (_Float16)0.f;
  }
  for (int i = tid; i < 32 * 64; i += 128) {
    int n = i / 64, k = i % 64;
    Wt2[i] = (_Float16)W2[k * 32 + n];
  }
  for (int i = tid; i < 64; i += 128) b1s[i] = B1[i];
  for (int i = tid; i < 32; i += 128) b2s[i] = B2[i];
  __syncthreads();

  const int wid = tid >> 5, lane = tid & 31;
  const int col = lane & 15, mrow = (lane >> 4) * 8;
  const int t = blockIdx.x;
  _Float16* xs = Xs[wid];
  _Float16* hs = Hs[wid];

  // Hoist loop-invariant B operands (8 x v16h = 64 VGPRs) and bias C-tiles
  // (6 x v8f = 48 VGPRs) into registers.
  v16h b1m[4], b2m[4];
  v8f cb1[4], cb2[2];
#pragma unroll
  for (int jn = 0; jn < 4; ++jn) {
    b1m[jn] = wmma_load_b(Wt1, 32, jn, lane, 0);
    cb1[jn] = splat8(b1s[jn * 16 + col]);
  }
#pragma unroll
  for (int jn = 0; jn < 2; ++jn) {
    b2m[jn * 2 + 0] = wmma_load_b(Wt2, 64, jn, lane, 0);
    b2m[jn * 2 + 1] = wmma_load_b(Wt2, 64, jn, lane, 32);
    cb2[jn] = splat8(b2s[jn * 16 + col]);
  }
  // Pre-zero the K-pad columns (16..31) of the X tile once.
  for (int i = lane; i < 256; i += 32)
    xs[(i >> 4) * 32 + 16 + (i & 15)] = (_Float16)0.f;

  const char* gbase =
      (const char*)(PF + (size_t)t * 1024 * 16) + (size_t)lane * 16;
  char* l0 = (char*)&Xf[wid][0][0] + lane * 16;
  char* l1 = (char*)&Xf[wid][1][0] + lane * 16;

#if ASYNC_LDS
  {  // prologue: stage first tile (sb = wid); 2 x 512B per tile
    const char* g = gbase + (size_t)wid * 1024;
    async_b128<0>(g, l0);
    async_b128<512>(g, l0);
  }
#endif
  int buf = 0;
  for (int sb = wid; sb < 64; sb += 4) {
    const int s0 = sb * 16;
#if ASYNC_LDS
    if (sb + 4 < 64) {
      const char* g = gbase + (size_t)(sb + 4) * 1024;
      char* ln = buf ? l0 : l1;
      async_b128<0>(g, ln);
      async_b128<512>(g, ln);
    }
    WAIT_ASYNC(2);  // current buffer's 2 copies are complete
    const float* src = Xf[wid][buf];
    for (int i = lane; i < 256; i += 32)
      xs[(i >> 4) * 32 + (i & 15)] = (_Float16)src[i];
    buf ^= 1;
#else
    const float* xb = PF + ((size_t)t * 1024 + s0) * 16;
    for (int i = lane; i < 256; i += 32)
      xs[(i >> 4) * 32 + (i & 15)] = (_Float16)xb[i];
#endif
    wait_lds();
#pragma unroll
    for (int jn = 0; jn < 4; ++jn) {
      v8f c = wmma_f16(wmma_load_a(xs, 32, lane, 0), b1m[jn], cb1[jn]);
#pragma unroll
      for (int r = 0; r < 8; ++r)
        hs[(mrow + r) * 64 + jn * 16 + col] = (_Float16)fmaxf((float)c[r], 0.f);
    }
    wait_lds();
#pragma unroll
    for (int jn = 0; jn < 2; ++jn) {
      v8f c = wmma_f16(wmma_load_a(hs, 64, lane, 0), b2m[jn * 2 + 0], cb2[jn]);
      c = wmma_f16(wmma_load_a(hs, 64, lane, 32), b2m[jn * 2 + 1], c);
#pragma unroll
      for (int r = 0; r < 8; ++r)
        PE[((size_t)(t * 1024 + s0 + mrow + r)) * 32 + jn * 16 + col] =
            (_Float16)fmaxf((float)c[r], 0.f);
    }
  }
}

// ---------------------------------------------------------------------------
// Masked aggregations (small; p_emb read is L2-resident).
// ---------------------------------------------------------------------------
__global__ void k_aggr_task(const unsigned char* __restrict__ M,
                            const float* __restrict__ SU,  // s_emb cols 0-31
                            const _Float16* __restrict__ PE,
                            float* __restrict__ TU) {
  const int t = blockIdx.x, lane = threadIdx.x;  // 32 threads = E dims
  const _Float16* per = PE + (size_t)t * 1024 * 32;
  float acc = 0.f, cnt = 0.f;
  for (int s = 0; s < 1024; ++s) {
    float mv = M[(size_t)t * 1024 + s] ? 1.f : 0.f;
    acc += mv * (SU[(size_t)s * 64 + lane] + (float)per[(size_t)s * 32 + lane]);
    cnt += mv;
  }
  TU[(size_t)t * 64 + 32 + lane] = acc / fmaxf(cnt, 1.f);
}

__global__ void k_aggr_server(const unsigned char* __restrict__ M,
                              const float* __restrict__ TU,  // t_emb cols 0-31
                              const _Float16* __restrict__ PE,
                              float* __restrict__ SU) {
  const int s = blockIdx.x, lane = threadIdx.x;
  float acc = 0.f, cnt = 0.f;
  for (int t = 0; t < 2048; ++t) {
    float mv = M[(size_t)t * 1024 + s] ? 1.f : 0.f;
    acc += mv * (TU[(size_t)t * 64 + lane] +
                 (float)PE[((size_t)t * 1024 + s) * 32 + lane]);
    cnt += mv;
  }
  SU[(size_t)s * 64 + 32 + lane] = acc / fmaxf(cnt, 1.f);
}

// ---------------------------------------------------------------------------
// Final scoring, 16 tasks per block: a_s s-tile staged once in LDS (async,
// double buffered), then lifted into D-layout registers (v8f asv[4]) and
// reused across 16 tasks -> C operand built with VALU adds, not DS loads.
// WMMA C = a_t + a_s (D = A*B + C); w_p^T in registers; dot with sh_w2 via
// __shfl_xor (width 16).
// ---------------------------------------------------------------------------
__global__ void __launch_bounds__(128)
k_score(const _Float16* __restrict__ PE, const float* __restrict__ AT,
        const float* __restrict__ AS, const float* __restrict__ SH_W1,
        const float* __restrict__ SH_W2, const float* __restrict__ SH_B2,
        const unsigned char* __restrict__ M, float* __restrict__ OUT) {
  __shared__ __align__(32) _Float16 Wpt[64 * 32];  // w_p^T: [n][k]
  __shared__ float w2s[64];
  __shared__ float ATs[16 * 64];                   // 16 tasks x 64 (a_t+sh_b1)
  __shared__ __align__(16) float ASs[4][2][16 * 64];  // per-wave double buffer

  const int tid = threadIdx.x;
  const int t0 = blockIdx.x * 16;
  for (int i = tid; i < 64 * 32; i += 128) {
    int n = i / 32, k = i % 32;
    Wpt[i] = (_Float16)SH_W1[(size_t)(64 + k) * 64 + n];  // rows 64..95 = w_p
  }
  for (int i = tid; i < 64; i += 128) w2s[i] = SH_W2[i];
  for (int i = tid; i < 16 * 64; i += 128) ATs[i] = AT[(size_t)t0 * 64 + i];
  __syncthreads();

  const int wid = tid >> 5, lane = tid & 31;
  const int col = lane & 15, mrow = (lane >> 4) * 8;
  const float b2v = SH_B2[0];

  v16h bm[4];
  float w2v[4];
#pragma unroll
  for (int jn = 0; jn < 4; ++jn) {
    bm[jn] = wmma_load_b(Wpt, 32, jn, lane, 0);
    w2v[jn] = w2s[jn * 16 + col];
  }

  char* l0 = (char*)&ASs[wid][0][0] + lane * 16;
  char* l1 = (char*)&ASs[wid][1][0] + lane * 16;
#if ASYNC_LDS
  {  // prologue: stage a_s tile for sb = wid (16 x 64 f32 = 4KB = 8 x 512B)
    const char* g = (const char*)(AS + (size_t)wid * 16 * 64) + lane * 16;
    async_b128<0>(g, l0);    async_b128<512>(g, l0);
    async_b128<1024>(g, l0); async_b128<1536>(g, l0);
    async_b128<2048>(g, l0); async_b128<2560>(g, l0);
    async_b128<3072>(g, l0); async_b128<3584>(g, l0);
  }
#endif
  int buf = 0;
  for (int sb = wid; sb < 64; sb += 4) {
    const int s0 = sb * 16;
    const float* cur = ASs[wid][buf];
#if ASYNC_LDS
    if (sb + 4 < 64) {
      const char* g =
          (const char*)(AS + (size_t)(sb + 4) * 16 * 64) + lane * 16;
      char* ln = buf ? l0 : l1;
      async_b128<0>(g, ln);    async_b128<512>(g, ln);
      async_b128<1024>(g, ln); async_b128<1536>(g, ln);
      async_b128<2048>(g, ln); async_b128<2560>(g, ln);
      async_b128<3072>(g, ln); async_b128<3584>(g, ln);
    }
    WAIT_ASYNC(8);  // current buffer's 8 copies complete
#else
    float* curw = ASs[wid][buf];
    for (int i = lane; i < 16 * 64; i += 32) curw[i] = AS[(size_t)s0 * 64 + i];
    wait_lds();
#endif
    // Lift the a_s tile into D-layout registers once per s-tile.
    v8f asv[4];
#pragma unroll
    for (int jn = 0; jn < 4; ++jn)
#pragma unroll
      for (int r = 0; r < 8; ++r)
        asv[jn][r] = cur[(mrow + r) * 64 + jn * 16 + col];

    for (int ti = 0; ti < 16; ++ti) {
      const int t = t0 + ti;
      v16h a = wmma_load_a(PE + ((size_t)t * 1024 + s0) * 32, 32, lane, 0);
      v8f d[4];
#pragma unroll
      for (int jn = 0; jn < 4; ++jn) {
        const float atv = ATs[ti * 64 + jn * 16 + col];
        v8f c;
#pragma unroll
        for (int r = 0; r < 8; ++r) c[r] = atv + asv[jn][r];
        d[jn] = wmma_f16(a, bm[jn], c);
      }
#pragma unroll
      for (int r = 0; r < 8; ++r) {
        float acc = 0.f;
#pragma unroll
        for (int jn = 0; jn < 4; ++jn)
          acc += fmaxf((float)d[jn][r], 0.f) * w2v[jn];
        acc += __shfl_xor(acc, 1, 16);
        acc += __shfl_xor(acc, 2, 16);
        acc += __shfl_xor(acc, 4, 16);
        acc += __shfl_xor(acc, 8, 16);
        if (col == 0) {
          const size_t idx = (size_t)t * 1024 + s0 + mrow + r;
          OUT[idx] = M[idx] ? (acc + b2v) : -1.0e9f;
        }
      }
    }
    buf ^= 1;
  }
}

// ---------------------------------------------------------------------------
extern "C" void kernel_launch(void* const* d_in, const int* in_sizes, int n_in,
                              void* d_out, int out_size, void* d_ws,
                              size_t ws_size, hipStream_t stream) {
  (void)in_sizes; (void)n_in; (void)out_size; (void)ws_size;
  const float* task_feat   = (const float*)d_in[0];
  const float* server_feat = (const float*)d_in[1];
  const float* pair_feat   = (const float*)d_in[2];
  const unsigned char* mask = (const unsigned char*)d_in[3];
  const float* te_w1 = (const float*)d_in[4];
  const float* te_b1 = (const float*)d_in[5];
  const float* te_w2 = (const float*)d_in[6];
  const float* te_b2 = (const float*)d_in[7];
  const float* se_w1 = (const float*)d_in[8];
  const float* se_b1 = (const float*)d_in[9];
  const float* se_w2 = (const float*)d_in[10];
  const float* se_b2 = (const float*)d_in[11];
  const float* pe_w1 = (const float*)d_in[12];
  const float* pe_b1 = (const float*)d_in[13];
  const float* pe_w2 = (const float*)d_in[14];
  const float* pe_b2 = (const float*)d_in[15];
  const float* tu_w1 = (const float*)d_in[16];
  const float* tu_b1 = (const float*)d_in[17];
  const float* tu_w2 = (const float*)d_in[18];
  const float* tu_b2 = (const float*)d_in[19];
  const float* su_w1 = (const float*)d_in[20];
  const float* su_b1 = (const float*)d_in[21];
  const float* su_w2 = (const float*)d_in[22];
  const float* su_b2 = (const float*)d_in[23];
  const float* sh_w1 = (const float*)d_in[24];
  const float* sh_b1 = (const float*)d_in[25];
  const float* sh_w2 = (const float*)d_in[26];
  const float* sh_b2 = (const float*)d_in[27];

  // Workspace layout: p_emb f16 (128 MB, L2-resident) + small f32 buffers.
  char* ws = (char*)d_ws;
  _Float16* pe   = (_Float16*)ws;                                 // 2048*1024*32 f16
  float* tu_in   = (float*)(ws + (size_t)2048 * 1024 * 32 * 2);   // 2048 x 64
  float* su_in   = tu_in + (size_t)2048 * 64;                     // 1024 x 64
  float* a_t     = su_in + (size_t)1024 * 64;                     // 2048 x 64
  float* a_s     = a_t + (size_t)2048 * 64;                       // 1024 x 64

  // 1) t_emb, s_emb into cols 0-31 of the concat buffers.
  k_mlp<32, false><<<32, 128, 0, stream>>>(task_feat, 32, 2048, te_w1, te_b1,
                                           te_w2, te_b2, nullptr, nullptr,
                                           tu_in, 64);
  k_mlp<32, false><<<16, 128, 0, stream>>>(server_feat, 32, 1024, se_w1, se_b1,
                                           se_w2, se_b2, nullptr, nullptr,
                                           su_in, 64);
  // 2) p_emb (the big pass), f16 to workspace.
  k_pair1<<<2048, 128, 0, stream>>>(pair_feat, pe_w1, pe_b1, pe_w2, pe_b2, pe);
  // 3) masked aggregations into cols 32-63.
  k_aggr_task<<<2048, 32, 0, stream>>>(mask, su_in, pe, tu_in);
  k_aggr_server<<<1024, 32, 0, stream>>>(mask, tu_in, pe, su_in);
  // 4) fused update MLP + projection: a_t = MLP(tu_in)@w_t + sh_b1, a_s = ...
  k_mlp<64, true><<<32, 128, 0, stream>>>(tu_in, 64, 2048, tu_w1, tu_b1, tu_w2,
                                          tu_b2, sh_w1, sh_b1, a_t, 64);
  k_mlp<64, true><<<16, 128, 0, stream>>>(su_in, 64, 1024, su_w1, su_b1, su_w2,
                                          su_b2, sh_w1 + 32 * 64, nullptr,
                                          a_s, 64);
  // 5) final scoring.
  k_score<<<128, 128, 0, stream>>>(pe, a_t, a_s, sh_w1, sh_w2, sh_b2, mask,
                                   (float*)d_out);
}
